// GCNBMPEncoder_84756884619770
// MI455X (gfx1250) — matrix operand order repaired and checked by reading, not compiled, not run on hardware
//
#include <hip/hip_runtime.h>
#include <hip/hip_bf16.h>

#define NNODES 50000
#define NEDGES 800000
#define DIN    128
#define DH     256
#define NREL   4
#define EPSV   1e-10f

typedef __attribute__((ext_vector_type(16))) __bf16 v16bf;
typedef __attribute__((ext_vector_type(8)))  float  v8f;

union Frag32B { v16bf v; uint4 q[2]; };

__device__ __forceinline__ unsigned short f2bf(float f) {
  union { float f; unsigned int u; } c; c.f = f;
  unsigned int u = c.u;
  u += 0x7FFFu + ((u >> 16) & 1u);   // round-to-nearest-even
  return (unsigned short)(u >> 16);
}

// ---------------------------------------------------------------- utilities
__global__ void zero_f32_kernel(float* __restrict__ p, long n) {
  long i = (long)blockIdx.x * blockDim.x + threadIdx.x;
  if (i < n) p[i] = 0.0f;
}

__global__ void add_bias_kernel(const float* __restrict__ a,
                                const float* __restrict__ b,
                                float* __restrict__ o) {
  int i = threadIdx.x;
  o[i] = a[i] + b[i];
}

// Pack [rel_w ; loop_w] (row-major [K,256] f32) -> column-major bf16 [256][Ktot]
__global__ void pack_cat_w_kernel(const float* __restrict__ rw,
                                  const float* __restrict__ lw,
                                  unsigned short* __restrict__ W,
                                  int Dself, int Ktot) {
  long t = (long)blockIdx.x * blockDim.x + threadIdx.x;
  long total = (long)DH * Ktot;
  if (t >= total) return;
  int n = (int)(t / Ktot);
  int k = (int)(t % Ktot);
  int Krel = Ktot - Dself;
  float v = (k < Krel) ? rw[(size_t)k * DH + n] : lw[(size_t)(k - Krel) * DH + n];
  W[(size_t)n * Ktot + k] = f2bf(v);
}

// Transpose 256x256 f32 row-major -> column-major bf16 [256][256]
__global__ void pack_sq_w_kernel(const float* __restrict__ w,
                                 unsigned short* __restrict__ W) {
  int t = blockIdx.x * blockDim.x + threadIdx.x;
  if (t >= DH * DH) return;
  int n = t / DH, k = t % DH;
  W[n * DH + k] = f2bf(w[k * DH + n]);
}

// ---------------------------------------------------------------- scatter-mean
__global__ void scatter_kernel(const float* __restrict__ x,
                               const int* __restrict__ src,
                               const int* __restrict__ dst,
                               const int* __restrict__ rel,
                               float* __restrict__ agg,
                               float* __restrict__ cnt, int D) {
  long t = (long)blockIdx.x * blockDim.x + threadIdx.x;
  int chunks = D >> 2;
  long total = (long)NEDGES * chunks;
  if (t >= total) return;
  int e = (int)(t / chunks);
  int c = (int)(t % chunks) << 2;
  int s = src[e], d = dst[e], r = rel[e];
  float4 v = *(const float4*)(x + (size_t)s * D + c);
  float* a = agg + ((size_t)d * NREL + r) * D + c;
  atomicAdd(a + 0, v.x);
  atomicAdd(a + 1, v.y);
  atomicAdd(a + 2, v.z);
  atomicAdd(a + 3, v.w);
  if (c == 0) atomicAdd(cnt + d * NREL + r, 1.0f);
}

// Normalize agg by count, concat self features, convert to bf16 row-major [N, Ktot]
__global__ void build_a_kernel(const float* __restrict__ agg,
                               const float* __restrict__ cnt,
                               const float* __restrict__ self,
                               unsigned short* __restrict__ A,
                               int D, int Ktot) {
  long t = (long)blockIdx.x * blockDim.x + threadIdx.x;
  long total = (long)NNODES * Ktot;
  if (t >= total) return;
  int row = (int)(t / Ktot);
  int k = (int)(t % Ktot);
  int Krel = NREL * D;
  float v;
  if (k < Krel) {
    float c = cnt[row * NREL + (k / D)];
    v = agg[(size_t)row * Krel + k] / (c + EPSV);
  } else {
    v = self[(size_t)row * D + (k - Krel)];
  }
  A[(size_t)row * Ktot + k] = f2bf(v);
}

// ---------------------------------------------------------------- WMMA GEMMs
// out = relu(A[N,Ktot]bf16 @ Wcolmajor[256][Ktot]bf16 + bias); writes f32 + bf16
__global__ __launch_bounds__(128)
void rgc_gemm_kernel(const unsigned short* __restrict__ A,
                     const unsigned short* __restrict__ W,
                     const float* __restrict__ bias,
                     float* __restrict__ outF,
                     unsigned short* __restrict__ outBF,
                     int Ktot) {
  const int lane = threadIdx.x & 31;
  const int wave = threadIdx.x >> 5;
  const int m0 = blockIdx.x << 4;      // 16 rows per block
  const int n0 = wave << 6;            // 64 cols per wave
  const int l15 = lane & 15;
  const int hi = lane >> 4;            // 0: lanes 0-15, 1: lanes 16-31
  // 16-bit A layout: lanes 0-15 hold K {0..7, 16..23}, lanes 16-31 K {8..15, 24..31}
  const unsigned short* arow = A + (size_t)(m0 + l15) * Ktot + (hi ? 8 : 0);
  // 16-bit B layout: lanes 0-15 hold K {0..7, 16..23}, lanes 16-31 K {8..15, 24..31} (colmajor)
  const int kbB = hi ? 8 : 0;

  v8f acc[4] = {};
  for (int k0 = 0; k0 < Ktot; k0 += 32) {
    Frag32B a;
    a.q[0] = *(const uint4*)(arow + k0);
    a.q[1] = *(const uint4*)(arow + k0 + 16);
#pragma unroll
    for (int j = 0; j < 4; ++j) {
      const unsigned short* wp = W + (size_t)(n0 + j * 16 + l15) * Ktot + k0 + kbB;
      Frag32B b;
      b.q[0] = *(const uint4*)(wp);
      b.q[1] = *(const uint4*)(wp + 16);
      acc[j] = __builtin_amdgcn_wmma_f32_16x16x32_bf16(
          false, a.v, false, b.v, (short)0, acc[j], false, false);
    }
  }
#pragma unroll
  for (int j = 0; j < 4; ++j) {
    const int col = n0 + j * 16 + l15;
    const float bb = bias[col];
#pragma unroll
    for (int v = 0; v < 8; ++v) {
      const int row = m0 + v + (hi << 3);
      float val = fmaxf(acc[j][v] + bb, 0.0f);
      outF[(size_t)row * DH + col] = val;
      outBF[(size_t)row * DH + col] = f2bf(val);
    }
  }
}

// Highway: p = relu(H@Wp + bp), g = sigmoid(H@Wt + bt), out = g*p + (1-g)*H
__global__ __launch_bounds__(128)
void highway_gemm_kernel(const unsigned short* __restrict__ Hbf,
                         const float* __restrict__ Hf,
                         const unsigned short* __restrict__ Wp,
                         const float* __restrict__ bp,
                         const unsigned short* __restrict__ Wt,
                         const float* __restrict__ bt,
                         float* __restrict__ outF) {
  const int lane = threadIdx.x & 31;
  const int wave = threadIdx.x >> 5;
  const int m0 = blockIdx.x << 4;
  const int n0 = wave << 6;
  const int l15 = lane & 15;
  const int hi = lane >> 4;
  const unsigned short* arow = Hbf + (size_t)(m0 + l15) * DH + (hi ? 8 : 0);
  const int kbB = hi ? 8 : 0;

  v8f accP[4] = {};
  v8f accT[4] = {};
  for (int k0 = 0; k0 < DH; k0 += 32) {
    Frag32B a;
    a.q[0] = *(const uint4*)(arow + k0);
    a.q[1] = *(const uint4*)(arow + k0 + 16);
#pragma unroll
    for (int j = 0; j < 4; ++j) {
      const size_t wo = (size_t)(n0 + j * 16 + l15) * DH + k0 + kbB;
      Frag32B bP, bT;
      bP.q[0] = *(const uint4*)(Wp + wo);
      bP.q[1] = *(const uint4*)(Wp + wo + 16);
      bT.q[0] = *(const uint4*)(Wt + wo);
      bT.q[1] = *(const uint4*)(Wt + wo + 16);
      accP[j] = __builtin_amdgcn_wmma_f32_16x16x32_bf16(
          false, a.v, false, bP.v, (short)0, accP[j], false, false);
      accT[j] = __builtin_amdgcn_wmma_f32_16x16x32_bf16(
          false, a.v, false, bT.v, (short)0, accT[j], false, false);
    }
  }
#pragma unroll
  for (int j = 0; j < 4; ++j) {
    const int col = n0 + j * 16 + l15;
    const float bpc = bp[col], btc = bt[col];
#pragma unroll
    for (int v = 0; v < 8; ++v) {
      const int row = m0 + v + (hi << 3);
      float p = fmaxf(accP[j][v] + bpc, 0.0f);
      float g = 1.0f / (1.0f + __expf(-(accT[j][v] + btc)));
      float xv = Hf[(size_t)row * DH + col];
      outF[(size_t)row * DH + col] = g * p + (1.0f - g) * xv;
    }
  }
}

// ---------------------------------------------------------------- launch
extern "C" void kernel_launch(void* const* d_in, const int* in_sizes, int n_in,
                              void* d_out, int out_size, void* d_ws, size_t ws_size,
                              hipStream_t stream) {
  const float* x       = (const float*)d_in[0];
  const int*   src     = (const int*)d_in[1];
  const int*   dst     = (const int*)d_in[2];
  const int*   rel     = (const int*)d_in[3];
  const float* rel_w1  = (const float*)d_in[4];
  const float* rel_b1  = (const float*)d_in[5];
  const float* loop_w1 = (const float*)d_in[6];
  const float* loop_b1 = (const float*)d_in[7];
  const float* hp_w1   = (const float*)d_in[8];
  const float* hp_b1   = (const float*)d_in[9];
  const float* ht_w1   = (const float*)d_in[10];
  const float* ht_b1   = (const float*)d_in[11];
  const float* rel_w2  = (const float*)d_in[12];
  const float* rel_b2  = (const float*)d_in[13];
  const float* loop_w2 = (const float*)d_in[14];
  const float* loop_b2 = (const float*)d_in[15];
  const float* hp_w2   = (const float*)d_in[16];
  const float* hp_b2   = (const float*)d_in[17];
  const float* ht_w2   = (const float*)d_in[18];
  const float* ht_b2   = (const float*)d_in[19];

  const int K1 = NREL * DIN + DIN;   // 640
  const int K2 = NREL * DH + DH;     // 1280

  char* ws = (char*)d_ws;
  size_t off = 0;
  auto alloc = [&](size_t bytes) -> char* {
    char* p = ws + off;
    off = (off + bytes + 255) & ~(size_t)255;
    return p;
  };
  float*          agg  = (float*)alloc((size_t)NNODES * NREL * DH * 4);      // 205 MB (max layer)
  float*          cnt  = (float*)alloc((size_t)NNODES * NREL * 4);
  unsigned short* Abuf = (unsigned short*)alloc((size_t)NNODES * K2 * 2);    // 128 MB
  float*          Hf   = (float*)alloc((size_t)NNODES * DH * 4);
  unsigned short* Hbf  = (unsigned short*)alloc((size_t)NNODES * DH * 2);
  float*          Gf   = (float*)alloc((size_t)NNODES * DH * 4);
  unsigned short* W1   = (unsigned short*)alloc((size_t)K1 * DH * 2);
  unsigned short* W2   = (unsigned short*)alloc((size_t)K2 * DH * 2);
  unsigned short* Wp1  = (unsigned short*)alloc((size_t)DH * DH * 2);
  unsigned short* Wt1  = (unsigned short*)alloc((size_t)DH * DH * 2);
  unsigned short* Wp2  = (unsigned short*)alloc((size_t)DH * DH * 2);
  unsigned short* Wt2  = (unsigned short*)alloc((size_t)DH * DH * 2);
  float*          B1   = (float*)alloc(DH * 4);
  float*          B2   = (float*)alloc(DH * 4);

  auto blocks = [](long n) -> unsigned { return (unsigned)((n + 255) / 256); };

  // Pack weights (colmajor bf16) + combined biases
  pack_cat_w_kernel<<<blocks((long)DH * K1), 256, 0, stream>>>(rel_w1, loop_w1, W1, DIN, K1);
  pack_cat_w_kernel<<<blocks((long)DH * K2), 256, 0, stream>>>(rel_w2, loop_w2, W2, DH, K2);
  pack_sq_w_kernel<<<blocks(DH * DH), 256, 0, stream>>>(hp_w1, Wp1);
  pack_sq_w_kernel<<<blocks(DH * DH), 256, 0, stream>>>(ht_w1, Wt1);
  pack_sq_w_kernel<<<blocks(DH * DH), 256, 0, stream>>>(hp_w2, Wp2);
  pack_sq_w_kernel<<<blocks(DH * DH), 256, 0, stream>>>(ht_w2, Wt2);
  add_bias_kernel<<<1, DH, 0, stream>>>(rel_b1, loop_b1, B1);
  add_bias_kernel<<<1, DH, 0, stream>>>(rel_b2, loop_b2, B2);

  // ---------------- layer 1: RGC(128->256) + Highway ----------------
  {
    long na = (long)NNODES * NREL * DIN;
    zero_f32_kernel<<<blocks(na), 256, 0, stream>>>(agg, na);
    long nc = (long)NNODES * NREL;
    zero_f32_kernel<<<blocks(nc), 256, 0, stream>>>(cnt, nc);
    long ns = (long)NEDGES * (DIN / 4);
    scatter_kernel<<<blocks(ns), 256, 0, stream>>>(x, src, dst, rel, agg, cnt, DIN);
    long nb = (long)NNODES * K1;
    build_a_kernel<<<blocks(nb), 256, 0, stream>>>(agg, cnt, x, Abuf, DIN, K1);
    rgc_gemm_kernel<<<NNODES / 16, 128, 0, stream>>>(Abuf, W1, B1, Hf, Hbf, K1);
    highway_gemm_kernel<<<NNODES / 16, 128, 0, stream>>>(Hbf, Hf, Wp1, hp_b1, Wt1, ht_b1, Gf);
  }

  // ---------------- layer 2: RGC(256->256) + Highway ----------------
  {
    long na = (long)NNODES * NREL * DH;
    zero_f32_kernel<<<blocks(na), 256, 0, stream>>>(agg, na);
    long nc = (long)NNODES * NREL;
    zero_f32_kernel<<<blocks(nc), 256, 0, stream>>>(cnt, nc);
    long ns = (long)NEDGES * (DH / 4);
    scatter_kernel<<<blocks(ns), 256, 0, stream>>>(Gf, src, dst, rel, agg, cnt, DH);
    long nb = (long)NNODES * K2;
    build_a_kernel<<<blocks(nb), 256, 0, stream>>>(agg, cnt, Gf, Abuf, DH, K2);
    rgc_gemm_kernel<<<NNODES / 16, 128, 0, stream>>>(Abuf, W2, B2, Hf, Hbf, K2);
    highway_gemm_kernel<<<NNODES / 16, 128, 0, stream>>>(Hbf, Hf, Wp2, hp_b2, Wt2, ht_b2,
                                                         (float*)d_out);
  }
}
